// DepthSeparableConv2d_conv3_1_37572373905418
// MI455X (gfx1250) — matrix operand order repaired and checked by reading, not compiled, not run on hardware
//
#include <hip/hip_runtime.h>
#include <hip/hip_bf16.h>

typedef __attribute__((ext_vector_type(8)))  _Float16 v8h;
typedef __attribute__((ext_vector_type(16))) _Float16 v16h;
typedef __attribute__((ext_vector_type(8)))  float    v8f;

#define BATCH   64
#define CIN     128
#define COUT    256
#define HDIM    56
#define HW      3136          // 56*56
#define MROWS   (BATCH * HW)  // 200704
#define EPSBN   1e-5f
#define DW_THR  4.0f
#define PW_THR  0.001f
#define TM      256           // M rows per workgroup (8 waves x 32)

// ---------------------------------------------------------------------------
// Kernel 1: depthwise 3x3 (pad 1) + bias + BN + ReLU + per-(b,c) prune.
// One workgroup per (b,c) plane; plane staged in LDS; result written as f16
// in [M = b*HW + hw][K = c] row-major layout for the WMMA GEMM A-operand.
// (y is 51 MB f16 -> lives in the 192 MB L2 for the GEMM pass.)
// ---------------------------------------------------------------------------
__global__ __launch_bounds__(256) void dw_bn_relu_prune(
    const float* __restrict__ x,   const float* __restrict__ dww,
    const float* __restrict__ dwb, const float* __restrict__ g1,
    const float* __restrict__ b1,  const float* __restrict__ m1,
    const float* __restrict__ v1,  _Float16* __restrict__ Y)
{
    __shared__ float tile[HW];   // 12544 B input plane
    __shared__ float wmax[8];

    const int tid = threadIdx.x;
    const int bc  = blockIdx.x;
    const int b   = bc >> 7;          // / CIN
    const int c   = bc & (CIN - 1);

    const float* __restrict__ xp = x + (size_t)bc * HW;
    for (int i = tid; i < HW; i += 256) tile[i] = xp[i];

    float w[9];
#pragma unroll
    for (int j = 0; j < 9; ++j) w[j] = dww[c * 9 + j];
    const float alpha = g1[c] * rsqrtf(v1[c] + EPSBN);
    const float beta  = b1[c] - m1[c] * alpha + alpha * dwb[c];

    __syncthreads();

    float vals[13];
    float mx = 0.f;
#pragma unroll
    for (int it = 0; it < 13; ++it) {
        const int hw = tid + it * 256;
        float v = 0.f;
        if (hw < HW) {
            const int hy = hw / HDIM;
            const int hx = hw - hy * HDIM;
            float acc = 0.f;
#pragma unroll
            for (int dy = -1; dy <= 1; ++dy) {
                const int yy = hy + dy;
                if ((unsigned)yy < (unsigned)HDIM) {
#pragma unroll
                    for (int dx = -1; dx <= 1; ++dx) {
                        const int xx = hx + dx;
                        if ((unsigned)xx < (unsigned)HDIM)
                            acc = fmaf(w[(dy + 1) * 3 + (dx + 1)],
                                       tile[yy * HDIM + xx], acc);
                    }
                }
            }
            v = alpha * acc + beta;
            v = v > 0.f ? v : 0.f;      // ReLU (so max|.| == max)
            mx = fmaxf(mx, v);
        }
        vals[it] = v;
    }

    // wave32 max-reduce, then cross-wave via LDS
#pragma unroll
    for (int off = 16; off > 0; off >>= 1)
        mx = fmaxf(mx, __shfl_xor(mx, off, 32));
    if ((tid & 31) == 0) wmax[tid >> 5] = mx;
    __syncthreads();
    float bm = wmax[0];
#pragma unroll
    for (int i = 1; i < 8; ++i) bm = fmaxf(bm, wmax[i]);
    const float keep = (bm >= DW_THR) ? 1.f : 0.f;

#pragma unroll
    for (int it = 0; it < 13; ++it) {
        const int hw = tid + it * 256;
        if (hw < HW)
            Y[((size_t)b * HW + hw) * CIN + c] = (_Float16)(vals[it] * keep);
    }
}

// ---------------------------------------------------------------------------
// Kernel 2: pointwise 1x1 as WMMA GEMM: Z[m,n] = Y[m,k] * W[n,k]^T
//   M = 200704, N = 256, K = 128. f16 inputs, f32 accumulate.
//
// Weights in 64 KB LDS, fragment-contiguous (conflict-free ds_load_b128):
//   Bs[chunk][n][j], chunk = kb*2 + laneHalf, j = 0..15.
// Each wave owns TWO 16-row A tiles (32 M rows) so every B fragment feeds two
// WMMAs (8 wmma per 8 ds_loads per N-tile) with two independent acc chains.
// B fragments are double-buffered across N-tiles; nt loop fully unrolled
// (128 v_wmma inline) so buffers are register-renamed, and BN/bias constants
// stay in per-lane register arrays.
// ---------------------------------------------------------------------------
__global__ __launch_bounds__(256) void pw_wmma_gemm(
    const _Float16* __restrict__ Y, const float* __restrict__ pw,
    const float* __restrict__ pwb,  const float* __restrict__ g2,
    const float* __restrict__ b2,   const float* __restrict__ m2,
    const float* __restrict__ v2,   float* __restrict__ Z,
    unsigned* __restrict__ chmax)
{
    __shared__ _Float16 Bs[8 * COUT * 16];   // exactly 64 KB

    const int tid = threadIdx.x;

    // Stage + convert weights: one output row (n) per thread; write each
    // 16-half k-chunk contiguously at its fragment slot.
    {
        const int n = tid;
        const float4* __restrict__ src = (const float4*)(pw + (size_t)n * CIN);
#pragma unroll
        for (int ch = 0; ch < 8; ++ch) {         // k = (ch>>1)*32 + (ch&1)*8 + {0..7, 16..23}
            v16h t;
#pragma unroll
            for (int q = 0; q < 4; ++q) {
                const float4 f = src[(ch >> 1) * 8 + (ch & 1) * 2 + (q >> 1) * 4 + (q & 1)];
                t[4 * q + 0] = (_Float16)f.x;
                t[4 * q + 1] = (_Float16)f.y;
                t[4 * q + 2] = (_Float16)f.z;
                t[4 * q + 3] = (_Float16)f.w;
            }
            *(v16h*)(Bs + ((size_t)ch * COUT + n) * 16) = t;
        }
    }
    __syncthreads();

    const int wave   = tid >> 5;
    const int lane   = tid & 31;
    const int laneHi = lane >> 4;     // 0: lanes 0-15, 1: lanes 16-31
    const int lmod   = lane & 15;
    const int mbase  = blockIdx.x * TM + wave * 32;

    // Hoist BN/bias epilogue constants: lane handles column n = nt*16 + lmod.
    float sc[16], bi[16];
#pragma unroll
    for (int nt = 0; nt < COUT / 16; ++nt) {
        const int n   = nt * 16 + lmod;
        const float s = g2[n] * rsqrtf(v2[n] + EPSBN);
        sc[nt] = s;
        bi[nt] = b2[n] - m2[n] * s + s * pwb[n];
    }

    // A fragments (16x32 f16 each): ISA layout -> lane L%16 holds row M=L%16;
    // low half-wave K {0..7,16..23}, high half-wave K {8..15,24..31} per step.
    const _Float16* __restrict__ Arow0 = Y + (size_t)(mbase + lmod) * CIN;
    const _Float16* __restrict__ Arow1 = Y + (size_t)(mbase + 16 + lmod) * CIN;
    __builtin_prefetch(Arow0, 0, 1);
    __builtin_prefetch(Arow1, 0, 1);
    v16h A0[4], A1[4];
#pragma unroll
    for (int kb = 0; kb < 4; ++kb) {
        const int ko = kb * 32 + laneHi * 8;
        const v8h lo0 = *(const v8h*)(Arow0 + ko);
        const v8h hi0 = *(const v8h*)(Arow0 + ko + 16);
        const v8h lo1 = *(const v8h*)(Arow1 + ko);
        const v8h hi1 = *(const v8h*)(Arow1 + ko + 16);
#pragma unroll
        for (int i = 0; i < 8; ++i) {
            A0[kb][i] = lo0[i]; A0[kb][i + 8] = hi0[i];
            A1[kb][i] = lo1[i]; A1[kb][i + 8] = hi1[i];
        }
    }

    // Each lane's 8 D rows are 8 consecutive m (HW % 8 == 0 -> same batch b)
    const int m0a  = mbase + laneHi * 8;
    const int bIa  = m0a / HW;
    const int hw0a = m0a - bIa * HW;
    const int m0b  = mbase + 16 + laneHi * 8;
    const int bIb  = m0b / HW;
    const int hw0b = m0b - bIb * HW;

    // Double-buffered B fragments across N-tiles.
    v16h Bb[2][4];
#pragma unroll
    for (int kb = 0; kb < 4; ++kb)
        Bb[0][kb] = *(const v16h*)(
            Bs + ((size_t)(kb * 2 + laneHi) * COUT + lmod) * 16);

#pragma unroll
    for (int nt = 0; nt < COUT / 16; ++nt) {
        const int n = nt * 16 + lmod;
        const int cur = nt & 1, nxt = cur ^ 1;

        // prefetch next tile's B fragments behind this tile's WMMA chain
        if (nt + 1 < COUT / 16) {
            const int n2 = (nt + 1) * 16 + lmod;
#pragma unroll
            for (int kb = 0; kb < 4; ++kb)
                Bb[nxt][kb] = *(const v16h*)(
                    Bs + ((size_t)(kb * 2 + laneHi) * COUT + n2) * 16);
        }

        v8f acc0 = {}, acc1 = {};
#pragma unroll
        for (int kb = 0; kb < 4; ++kb) {
            acc0 = __builtin_amdgcn_wmma_f32_16x16x32_f16(
                       false, A0[kb], false, Bb[cur][kb], (short)0, acc0,
                       false, false);
            acc1 = __builtin_amdgcn_wmma_f32_16x16x32_f16(
                       false, A1[kb], false, Bb[cur][kb], (short)0, acc1,
                       false, false);
        }

        // fused bias + BN + ReLU epilogue (constants already in registers)
        float oa[8], ob[8];
        float mxa = 0.f, mxb = 0.f;
#pragma unroll
        for (int r = 0; r < 8; ++r) {
            float va = fmaf(sc[nt], acc0[r], bi[nt]);
            va = va > 0.f ? va : 0.f;
            mxa = fmaxf(mxa, va);
            oa[r] = va;
            float vb = fmaf(sc[nt], acc1[r], bi[nt]);
            vb = vb > 0.f ? vb : 0.f;
            mxb = fmaxf(mxb, vb);
            ob[r] = vb;
        }

        float* __restrict__ zpa = Z + ((size_t)bIa * COUT + n) * HW + hw0a;
        *(float4*)(zpa)     = make_float4(oa[0], oa[1], oa[2], oa[3]);
        *(float4*)(zpa + 4) = make_float4(oa[4], oa[5], oa[6], oa[7]);
        atomicMax(chmax + (size_t)bIa * COUT + n, __float_as_uint(mxa));

        float* __restrict__ zpb = Z + ((size_t)bIb * COUT + n) * HW + hw0b;
        *(float4*)(zpb)     = make_float4(ob[0], ob[1], ob[2], ob[3]);
        *(float4*)(zpb + 4) = make_float4(ob[4], ob[5], ob[6], ob[7]);
        atomicMax(chmax + (size_t)bIb * COUT + n, __float_as_uint(mxb));
    }
}

// ---------------------------------------------------------------------------
// Kernel 3: apply pointwise-branch prune — zero a (b,o) slice iff its max < thr.
// ---------------------------------------------------------------------------
__global__ __launch_bounds__(256) void pw_prune_apply(
    const unsigned* __restrict__ chmax, float* __restrict__ Z)
{
    const int bo = blockIdx.x;
    if (__uint_as_float(chmax[bo]) >= PW_THR) return;
    float4* __restrict__ zp = (float4*)(Z + (size_t)bo * HW);
    const int tid = threadIdx.x;
#pragma unroll
    for (int i = 0; i < 4; ++i) {
        const int idx = tid + i * 256;
        if (idx < HW / 4) zp[idx] = make_float4(0.f, 0.f, 0.f, 0.f);
    }
}

// ---------------------------------------------------------------------------
extern "C" void kernel_launch(void* const* d_in, const int* in_sizes, int n_in,
                              void* d_out, int out_size, void* d_ws, size_t ws_size,
                              hipStream_t stream)
{
    (void)in_sizes; (void)n_in; (void)out_size; (void)ws_size;

    const float* x   = (const float*)d_in[0];
    const float* dww = (const float*)d_in[1];
    const float* dwb = (const float*)d_in[2];
    const float* g1  = (const float*)d_in[3];
    const float* b1  = (const float*)d_in[4];
    const float* m1  = (const float*)d_in[5];
    const float* v1  = (const float*)d_in[6];
    const float* pw  = (const float*)d_in[7];
    const float* pwb = (const float*)d_in[8];
    const float* g2  = (const float*)d_in[9];
    const float* b2  = (const float*)d_in[10];
    const float* m2  = (const float*)d_in[11];
    const float* v2  = (const float*)d_in[12];

    float*    Z = (float*)d_out;
    _Float16* Y = (_Float16*)d_ws;                                   // 51.4 MB
    unsigned* chmax =
        (unsigned*)((char*)d_ws + (size_t)MROWS * CIN * sizeof(_Float16));

    hipMemsetAsync(chmax, 0, (size_t)BATCH * COUT * sizeof(unsigned), stream);

    dw_bn_relu_prune<<<BATCH * CIN, 256, 0, stream>>>(x, dww, dwb, g1, b1, m1, v1, Y);
    pw_wmma_gemm<<<MROWS / TM, 256, 0, stream>>>(Y, pw, pwb, g2, b2, m2, v2, Z, chmax);
    pw_prune_apply<<<BATCH * COUT, 256, 0, stream>>>(chmax, Z);
}
